// DeepMDsimpleEnergyForcesSpecies_11854109737453
// MI455X (gfx1250) — matrix operand order repaired and checked by reading, not compiled, not run on hardware
//
#include <hip/hip_runtime.h>
#include <hip/hip_bf16.h>

typedef __attribute__((ext_vector_type(16))) _Float16 v16h;
typedef __attribute__((ext_vector_type(4)))  _Float16 v4h;
typedef __attribute__((ext_vector_type(8)))  float    v8f;

#define NSNAP 8
#define NPTS  4096
#define KNB   64
#define RCS   2.0f
#define RC    3.0f
#define PI_F  3.14159265358979f

#define WMMA_F16(A, B, C) \
    __builtin_amdgcn_wmma_f32_16x16x32_f16(false, (A), false, (B), (short)0, (C), false, false)

// ---------------------------------------------------------------------------
// Packed-weight fragment slots (each slot = 32 lanes x 16 halfs = 1 KB).
// ---------------------------------------------------------------------------
#define FR_FIT_W1 0            // kt 0..7, nt 0..3  -> slot + kt*4 + nt   (32)
#define FR_FIT_W2 32           // kt 0..1, nt 0..1  -> slot + kt*2 + nt   (4)
#define FR_FIT_W3 36           // 1
#define FR_FIT_W4 37           // 1
#define FR_EN_W0  38           // 1
#define FR_EN_W1  39           // 1
#define FR_EN_W2  40           // nt 0..1 (2)
#define FR_NSLOT  42

// B fragment built from row-major f32 weights W[K x N]; clamp+select, no branches.
// B layout: lane L holds col N=L%16; lanes 0-15 cover K 0..15 of the K-tile,
// lanes 16-31 cover K 16..31; VGPR v holds K = base+2v, base+2v+1.
__device__ inline v16h pack_b(const float* W, int K, int N, int kt, int nt, int lane)
{
    v16h b;
    const int n     = nt * 16 + (lane & 15);
    const int kbase = kt * 32 + ((lane & 16) ? 16 : 0);
    #pragma unroll
    for (int v = 0; v < 8; ++v) {
        #pragma unroll
        for (int h = 0; h < 2; ++h) {
            const int k  = kbase + 2 * v + h;
            const int kc = (k < K) ? k : (K - 1);          // clamped, always in-range
            const float w = W[kc * N + (n & (N - 1))];     // N is a power of two
            b[2 * v + h] = (k < K && n < N) ? (_Float16)w : (_Float16)0.f;
        }
    }
    return b;
}

// One aligned 32-byte load per B fragment.
__device__ inline v16h load_frag(const _Float16* pack, int slot, int lane)
{
    return *(const v16h*)(pack + ((size_t)slot * 32 + lane) * 16);
}

// A fragment: lane L holds row M=L%16; one contiguous 16-half run at
// K = kt*32 + (L<16?0:16) + {0..15}. Caller guarantees the row pitch
// covers kt*32+32 (padded with zeros) -> no guards, pure b128 loads.
__device__ inline v16h frag_a(const _Float16* X, int ld, int kt, int lane)
{
    const int m    = lane & 15;
    const int koff = (lane & 16) ? 16 : 0;
    const _Float16* p = X + m * ld + kt * 32 + koff;
    v16h a;
    #pragma unroll
    for (int e = 0; e < 8; ++e) a[e] = p[e];
    #pragma unroll
    for (int e = 0; e < 8; ++e) a[8 + e] = p[8 + e];
    return a;
}

// B fragment from row-major f16 LDS array X[K x ld]; caller guarantees
// k < K and n < ld are always in range (padded) -> no guards.
__device__ inline v16h frag_b_lds(const _Float16* X, int ld, int kt, int lane)
{
    const int n     = lane & 15;
    const int kbase = kt * 32 + ((lane & 16) ? 16 : 0);
    v16h b;
    #pragma unroll
    for (int v = 0; v < 8; ++v) {
        b[2 * v]     = X[(kbase + 2 * v)     * ld + n];
        b[2 * v + 1] = X[(kbase + 2 * v + 1) * ld + n];
    }
    return b;
}

// D accumulator -> f16 row-major Y[16 x ld]; writes the full 16-col tile,
// padded columns (n >= N) get 0 via select (no divergent skip). N power of 2.
__device__ inline void store_d_f16(const v8f& c, _Float16* Y, int ld, int nt,
                                   const float* bias, int N, bool relu, int lane)
{
    const int n  = nt * 16 + (lane & 15);
    const int mo = (lane & 16) ? 8 : 0;
    const float bv = bias[n & (N - 1)];
    #pragma unroll
    for (int j = 0; j < 8; ++j) {
        float y = c[j] + bv;
        if (relu) y = fmaxf(y, 0.f);
        y = (n < N) ? y : 0.f;
        Y[(j + mo) * ld + n] = (_Float16)y;
    }
}

// ---------------------------------------------------------------------------
// Kernel 1: zero outputs, build the 4-entry type-pair table, and pre-pack all
// WMMA B fragments (f16, lane order) so hot kernels load them with one b128x2.
// ---------------------------------------------------------------------------
__global__ __launch_bounds__(32) void k_prepare(
    const float* __restrict__ esW0, const float* __restrict__ esB0,
    const float* __restrict__ esW1, const float* __restrict__ esB1,
    const float* __restrict__ fsW0, const float* __restrict__ fsB0,
    const float* __restrict__ fsW1, const float* __restrict__ fsB1,
    const float* __restrict__ enW0, const float* __restrict__ enW1,
    const float* __restrict__ enW2,
    const float* __restrict__ fW1, const float* __restrict__ fW2,
    const float* __restrict__ fW3, const float* __restrict__ fW4,
    float* __restrict__ td_table, _Float16* __restrict__ pack,
    float* __restrict__ out)
{
    const int t = threadIdx.x;
    if (t < NSNAP) out[t] = 0.0f;

    // ---- type-pair table (types in {0,1} -> 4 entries) --------------------
    if (t < 4) {
        float p0 = (float)(t >> 1);
        float p1 = (float)(t & 1);
        float acc[4] = {0.f, 0.f, 0.f, 0.f};
        for (int pass = 0; pass < 2; ++pass) {
            float a = pass ? p1 : p0;
            float b = pass ? p0 : p1;
            float h[8];
            for (int o = 0; o < 8; ++o)
                h[o] = fmaxf(a * esW0[o] + b * esW0[8 + o] + esB0[o], 0.f);
            for (int j = 0; j < 4; ++j) {
                float v = esB1[j];
                for (int o = 0; o < 8; ++o) v += h[o] * esW1[o * 4 + j];
                acc[j] += v;
            }
        }
        float h2[4];
        for (int o = 0; o < 4; ++o) {
            float v = fsB0[o];
            for (int i = 0; i < 4; ++i) v += acc[i] * fsW0[i * 4 + o];
            h2[o] = fmaxf(v, 0.f);
        }
        for (int j = 0; j < 4; ++j) {
            float v = fsB1[j];
            for (int o = 0; o < 4; ++o) v += h2[o] * fsW1[o * 4 + j];
            td_table[t * 4 + j] = v;
        }
    }

    // ---- pack B fragments (each lane writes its own 16 halfs per slot) ----
    const int lane = t;
    #define PUT(slot, frag) *(v16h*)(pack + ((size_t)(slot) * 32 + lane) * 16) = (frag)

    for (int kt = 0; kt < 8; ++kt)
        for (int nt = 0; nt < 4; ++nt)
            PUT(FR_FIT_W1 + kt * 4 + nt, pack_b(fW1, 256, 64, kt, nt, lane));
    for (int kt = 0; kt < 2; ++kt)
        for (int nt = 0; nt < 2; ++nt)
            PUT(FR_FIT_W2 + kt * 2 + nt, pack_b(fW2, 64, 32, kt, nt, lane));
    PUT(FR_FIT_W3, pack_b(fW3, 32, 16, 0, 0, lane));
    PUT(FR_FIT_W4, pack_b(fW4, 16,  8, 0, 0, lane));
    PUT(FR_EN_W0,  pack_b(enW0, 4,  8, 0, 0, lane));
    PUT(FR_EN_W1,  pack_b(enW1, 8, 16, 0, 0, lane));
    PUT(FR_EN_W2 + 0, pack_b(enW2, 16, 32, 0, 0, lane));
    PUT(FR_EN_W2 + 1, pack_b(enW2, 16, 32, 0, 1, lane));
    #undef PUT
}

// ---------------------------------------------------------------------------
// Kernel 2: one 64-thread workgroup (2 waves) per atom; thread == neighbor.
// Geometry on VALU, then everything matrix-shaped on WMMA:
//   en MLP : [64x4]x[4x8] -> [64x8]x[8x16] -> [64x16]x[16x32]  (M = pairs)
//   R      : [4x64]x[64x8]     (K = pairs; rt^T zero-padded to 16 rows)
//   D      : [32x64]x[64x8]    (K = pairs; uses transposed G)
// All LDS activation buffers are padded so fragment loads need no guards.
// ---------------------------------------------------------------------------
__global__ __launch_bounds__(64) void k_atom_env(
    const float* __restrict__ pos, const int* __restrict__ types,
    const int* __restrict__ neigh,
    const float* __restrict__ enB0, const float* __restrict__ enB1,
    const float* __restrict__ enB2,
    const float* __restrict__ td_table, const _Float16* __restrict__ pack,
    const float* __restrict__ lenp,
    _Float16* __restrict__ Dflat)
{
    __shared__ _Float16 sX  [KNB * 32];   // s_tilde rows, cols 4..31 zero
    __shared__ _Float16 sH1 [KNB * 32];   // cols 8..31 zero
    __shared__ _Float16 sH2 [KNB * 32];   // cols 16..31 zero
    __shared__ _Float16 sG  [KNB * 32];   // G (B for R; all 32 cols real)
    __shared__ _Float16 sGt [32 * KNB];   // G^T (A for D)
    __shared__ _Float16 sRTt[16 * KNB];   // rt^T, rows 4..15 zero (A for R)
    __shared__ _Float16 sR1 [KNB * 16];   // cols 8..15 zero (B for D)
    __shared__ float    sRp [16 * 16];    // R accumulator dump (rows 0..3 used)
    __shared__ _Float16 sD  [32 * 16];    // D staging (cols 0..7 used)
    __shared__ float    sTd [16];

    const int t    = threadIdx.x;
    const int lane = t & 31;
    const int wave = t >> 5;
    const int atom = blockIdx.x & (NPTS - 1);
    const int s    = blockIdx.x >> 12;
    const int base = s * NPTS;

    // ---- zero-fill padded buffers (vectorized dword stores) --------------
    {
        uint32_t* z0 = (uint32_t*)sX;     // 1024 dwords
        uint32_t* z1 = (uint32_t*)sH1;    // 1024
        uint32_t* z2 = (uint32_t*)sH2;    // 1024
        uint32_t* z3 = (uint32_t*)sRTt;   // 512
        uint32_t* z4 = (uint32_t*)sR1;    // 512
        for (int i = t; i < 1024; i += 64) { z0[i] = 0u; z1[i] = 0u; z2[i] = 0u; }
        for (int i = t; i < 512;  i += 64) { z3[i] = 0u; z4[i] = 0u; }
    }
    if (t < 16) sTd[t] = td_table[t];
    __syncthreads();

    // ---- geometry (VALU) --------------------------------------------------
    const float length  = lenp[0];
    const float inv_len = 1.0f / length;

    const int  j      = neigh[(size_t)(base + atom) * KNB + t];
    const bool masked = (j < 0);
    const int  js     = masked ? 0 : j;

    const float xi = pos[(size_t)(base + atom) * 3 + 0];
    const float yi = pos[(size_t)(base + atom) * 3 + 1];
    const float zi = pos[(size_t)(base + atom) * 3 + 2];
    float dx = pos[(size_t)(base + js) * 3 + 0] - xi;
    float dy = pos[(size_t)(base + js) * 3 + 1] - yi;
    float dz = pos[(size_t)(base + js) * 3 + 2] - zi;
    dx -= length * rintf(dx * inv_len);
    dy -= length * rintf(dy * inv_len);
    dz -= length * rintf(dz * inv_len);

    const float r2    = dx * dx + dy * dy + dz * dz;
    const float r     = sqrtf(r2 + 1e-12f);
    const float inv_r = 1.0f / r;
    float sw;
    if (r < RCS)     sw = 1.0f;
    else if (r < RC) { float tt = (r - RCS) / (RC - RCS); sw = 0.5f * cosf(PI_F * tt) + 0.5f; }
    else             sw = 0.0f;

    const float s_ij = masked ? 0.0f : sw * inv_r;
    const float f    = s_ij * inv_r;
    const float rx = dx * f, ry = dy * f, rz = dz * f;

    const int ti = types[base + atom];
    const int tj = types[base + js];
    const float* td = sTd + (ti * 2 + tj) * 4;

    sX[t * 32 + 0] = (_Float16)(td[0] * s_ij);
    sX[t * 32 + 1] = (_Float16)(td[1] * s_ij);
    sX[t * 32 + 2] = (_Float16)(td[2] * s_ij);
    sX[t * 32 + 3] = (_Float16)(td[3] * s_ij);
    sRTt[0 * KNB + t] = (_Float16)s_ij;
    sRTt[1 * KNB + t] = (_Float16)rx;
    sRTt[2 * KNB + t] = (_Float16)ry;
    sRTt[3 * KNB + t] = (_Float16)rz;
    __syncthreads();

    // ---- en layer 1: [16x4] x [4x8] per M-tile ---------------------------
    #pragma unroll
    for (int mt = 0; mt < 2; ++mt) {
        const int row0 = (wave * 2 + mt) * 16;
        v8f c = {};
        c = WMMA_F16(frag_a(sX + row0 * 32, 32, 0, lane),
                     load_frag(pack, FR_EN_W0, lane), c);
        store_d_f16(c, sH1 + row0 * 32, 32, 0, enB0, 8, true, lane);
    }
    __syncthreads();

    // ---- en layer 2: [16x8] x [8x16] -------------------------------------
    #pragma unroll
    for (int mt = 0; mt < 2; ++mt) {
        const int row0 = (wave * 2 + mt) * 16;
        v8f c = {};
        c = WMMA_F16(frag_a(sH1 + row0 * 32, 32, 0, lane),
                     load_frag(pack, FR_EN_W1, lane), c);
        store_d_f16(c, sH2 + row0 * 32, 32, 0, enB1, 16, true, lane);
    }
    __syncthreads();

    // ---- en layer 3: [16x16] x [16x32], no relu; store G and G^T ---------
    #pragma unroll
    for (int mt = 0; mt < 2; ++mt) {
        const int row0 = (wave * 2 + mt) * 16;
        const v16h a = frag_a(sH2 + row0 * 32, 32, 0, lane);
        #pragma unroll
        for (int nt = 0; nt < 2; ++nt) {
            v8f c = {};
            c = WMMA_F16(a, load_frag(pack, FR_EN_W2 + nt, lane), c);
            const int n  = nt * 16 + (lane & 15);
            const int mo = (lane & 16) ? 8 : 0;
            const float bv = enB2[n];
            #pragma unroll
            for (int jj = 0; jj < 8; ++jj) {
                const int row = row0 + jj + mo;
                const _Float16 h = (_Float16)(c[jj] + bv);
                sG [row * 32 + n]  = h;
                sGt[n * KNB + row] = h;
            }
        }
    }
    __syncthreads();

    // ---- R = rt^T x G2 : [4x64] x [64x8], wave 0 only --------------------
    if (wave == 0) {
        v8f c = {};
        #pragma unroll
        for (int kt = 0; kt < 2; ++kt)
            c = WMMA_F16(frag_a(sRTt, KNB, kt, lane),
                         frag_b_lds(sG, 32, kt, lane), c);
        const int n  = lane & 15;
        const int mo = (lane & 16) ? 8 : 0;
        #pragma unroll
        for (int jj = 0; jj < 8; ++jj)
            sRp[(jj + mo) * 16 + n] = c[jj];      // rows >= 4 / cols >= 8 unused
    }
    __syncthreads();

    // ---- R1[k][e] = sum_d rt[k][d] * R[d][e]   (tiny, VALU) --------------
    #pragma unroll
    for (int e = 0; e < 8; ++e) {
        float v = 0.f;
        #pragma unroll
        for (int d = 0; d < 4; ++d) v += (float)sRTt[d * KNB + t] * sRp[d * 16 + e];
        sR1[t * 16 + e] = (_Float16)v;
    }
    __syncthreads();

    // ---- D = G^T x R1 : [32x64] x [64x8]; wave w -> rows 16w..16w+15 -----
    {
        v8f c = {};
        const _Float16* At = sGt + wave * 16 * KNB;
        #pragma unroll
        for (int kt = 0; kt < 2; ++kt)
            c = WMMA_F16(frag_a(At, KNB, kt, lane),
                         frag_b_lds(sR1, 16, kt, lane), c);
        const int n  = lane & 15;
        const int mo = (lane & 16) ? 8 : 0;
        #pragma unroll
        for (int jj = 0; jj < 8; ++jj)
            sD[(wave * 16 + jj + mo) * 16 + n] = (_Float16)c[jj];
    }
    __syncthreads();

    // ---- coalesced Dflat store: thread t writes halfs t*4 .. t*4+3 -------
    {
        _Float16* drow = Dflat + (size_t)(base + atom) * 256;
        v4h o;
        #pragma unroll
        for (int q = 0; q < 4; ++q) {
            const int idx = t * 4 + q;
            o[q] = sD[(idx >> 3) * 16 + (idx & 7)];
        }
        *(v4h*)(drow + t * 4) = o;
    }
}

// ---------------------------------------------------------------------------
// Kernel 3: fit network. One wave (32 lanes) per 16 atoms.
// Layers 256->64->32->16->8 via v_wmma_f32_16x16x32_f16; 8-wide tail layers
// and the per-type affine per-lane; per-snapshot reduction by atomicAdd.
// ---------------------------------------------------------------------------
__global__ __launch_bounds__(32) void k_fit(
    const _Float16* __restrict__ Dflat, const _Float16* __restrict__ pack,
    const float* __restrict__ b1, const float* __restrict__ b2,
    const float* __restrict__ b3, const float* __restrict__ b4,
    const float* __restrict__ W5, const float* __restrict__ b5,
    const float* __restrict__ W6, const float* __restrict__ b6,
    const float* __restrict__ W7, const float* __restrict__ b7,
    const float* __restrict__ W8, const float* __restrict__ b8,
    const int* __restrict__ types, const float* __restrict__ av,
    const float* __restrict__ stdv, float* __restrict__ out)
{
    __shared__ _Float16 Y1[16 * 64];
    __shared__ _Float16 Y2[16 * 32];
    __shared__ _Float16 Y3[16 * 32];     // cols 16..31 zero
    __shared__ float    Yf[16 * 16];     // cols 8..15 unused
    __shared__ float    part[16];

    const int lane  = threadIdx.x;
    const int gbase = blockIdx.x * 16;
    const int s     = gbase / NPTS;

    // zero the padded half of Y3 once
    {
        uint32_t* z = (uint32_t*)Y3;     // 256 dwords
        for (int i = lane; i < 256; i += 32) z[i] = 0u;
    }
    __syncthreads();

    // ---- layer 1: [16 x 256] x [256 x 64] -> 8 K-tiles, 4 N-tiles --------
    v8f acc0 = {}, acc1 = {}, acc2 = {}, acc3 = {};
    const _Float16* Arow = Dflat + (size_t)gbase * 256;
    for (int kt = 0; kt < 8; ++kt) {
        const v16h a = frag_a(Arow, 256, kt, lane);
        acc0 = WMMA_F16(a, load_frag(pack, FR_FIT_W1 + kt * 4 + 0, lane), acc0);
        acc1 = WMMA_F16(a, load_frag(pack, FR_FIT_W1 + kt * 4 + 1, lane), acc1);
        acc2 = WMMA_F16(a, load_frag(pack, FR_FIT_W1 + kt * 4 + 2, lane), acc2);
        acc3 = WMMA_F16(a, load_frag(pack, FR_FIT_W1 + kt * 4 + 3, lane), acc3);
    }
    store_d_f16(acc0, Y1, 64, 0, b1, 64, true, lane);
    store_d_f16(acc1, Y1, 64, 1, b1, 64, true, lane);
    store_d_f16(acc2, Y1, 64, 2, b1, 64, true, lane);
    store_d_f16(acc3, Y1, 64, 3, b1, 64, true, lane);
    __syncthreads();

    // ---- layer 2: [16 x 64] x [64 x 32] ----------------------------------
    v8f c0 = {}, c1 = {};
    #pragma unroll
    for (int kt = 0; kt < 2; ++kt) {
        const v16h a = frag_a(Y1, 64, kt, lane);
        c0 = WMMA_F16(a, load_frag(pack, FR_FIT_W2 + kt * 2 + 0, lane), c0);
        c1 = WMMA_F16(a, load_frag(pack, FR_FIT_W2 + kt * 2 + 1, lane), c1);
    }
    store_d_f16(c0, Y2, 32, 0, b2, 32, true, lane);
    store_d_f16(c1, Y2, 32, 1, b2, 32, true, lane);
    __syncthreads();

    // ---- layer 3: [16 x 32] x [32 x 16] ----------------------------------
    {
        v8f d0 = {};
        d0 = WMMA_F16(frag_a(Y2, 32, 0, lane), load_frag(pack, FR_FIT_W3, lane), d0);
        store_d_f16(d0, Y3, 32, 0, b3, 16, true, lane);
    }
    __syncthreads();

    // ---- layer 4: [16 x 16] x [16 x 8] (zero-padded K and N) -------------
    {
        v8f e0 = {};
        e0 = WMMA_F16(frag_a(Y3, 32, 0, lane), load_frag(pack, FR_FIT_W4, lane), e0);
        const int n  = lane & 15;
        const int mo = (lane & 16) ? 8 : 0;
        const float bv = b4[n & 7];
        #pragma unroll
        for (int j = 0; j < 8; ++j) {
            float y = fmaxf(e0[j] + bv, 0.f);
            Yf[(j + mo) * 16 + n] = (n < 8) ? y : 0.f;   // cols 8..15 ignored
        }
    }
    __syncthreads();

    // ---- tail: per-atom 8->8->8->8->1 and per-type affine ----------------
    if (lane < 16) {
        float y[8];
        #pragma unroll
        for (int i = 0; i < 8; ++i) y[i] = Yf[lane * 16 + i];

        #define DENSE8_RELU(Wl, Bl)                                      \
            do {                                                         \
                float z[8];                                              \
                _Pragma("unroll")                                        \
                for (int o = 0; o < 8; ++o) {                            \
                    float v = (Bl)[o];                                   \
                    _Pragma("unroll")                                    \
                    for (int i = 0; i < 8; ++i) v += y[i] * (Wl)[i*8+o]; \
                    z[o] = fmaxf(v, 0.f);                                \
                }                                                        \
                _Pragma("unroll")                                        \
                for (int i = 0; i < 8; ++i) y[i] = z[i];                 \
            } while (0)

        DENSE8_RELU(W5, b5);
        DENSE8_RELU(W6, b6);
        DENSE8_RELU(W7, b7);
        #undef DENSE8_RELU

        float e = b8[0];
        #pragma unroll
        for (int i = 0; i < 8; ++i) e += y[i] * W8[i];

        const int ty = types[gbase + lane];
        part[lane] = e * stdv[ty] + av[ty];
    }
    __syncthreads();

    if (lane == 0) {
        float sum = 0.f;
        #pragma unroll
        for (int i = 0; i < 16; ++i) sum += part[i];
        atomicAdd(&out[s], sum);
    }
}

// ---------------------------------------------------------------------------
// Host launcher
// ---------------------------------------------------------------------------
extern "C" void kernel_launch(void* const* d_in, const int* in_sizes, int n_in,
                              void* d_out, int out_size, void* d_ws, size_t ws_size,
                              hipStream_t stream)
{
    (void)in_sizes; (void)n_in; (void)out_size; (void)ws_size;

    const float* pos   = (const float*)d_in[0];
    const int*   types = (const int*)  d_in[1];
    const int*   neigh = (const int*)  d_in[2];

    const float* esW0 = (const float*)d_in[3];
    const float* esB0 = (const float*)d_in[4];
    const float* esW1 = (const float*)d_in[5];
    const float* esB1 = (const float*)d_in[6];

    const float* fsW0 = (const float*)d_in[7];
    const float* fsB0 = (const float*)d_in[8];
    const float* fsW1 = (const float*)d_in[9];
    const float* fsB1 = (const float*)d_in[10];

    const float* enW0 = (const float*)d_in[11];
    const float* enB0 = (const float*)d_in[12];
    const float* enW1 = (const float*)d_in[13];
    const float* enB1 = (const float*)d_in[14];
    const float* enW2 = (const float*)d_in[15];
    const float* enB2 = (const float*)d_in[16];

    const float* fW1 = (const float*)d_in[17]; const float* fb1 = (const float*)d_in[18];
    const float* fW2 = (const float*)d_in[19]; const float* fb2 = (const float*)d_in[20];
    const float* fW3 = (const float*)d_in[21]; const float* fb3 = (const float*)d_in[22];
    const float* fW4 = (const float*)d_in[23]; const float* fb4 = (const float*)d_in[24];
    const float* fW5 = (const float*)d_in[25]; const float* fb5 = (const float*)d_in[26];
    const float* fW6 = (const float*)d_in[27]; const float* fb6 = (const float*)d_in[28];
    const float* fW7 = (const float*)d_in[29]; const float* fb7 = (const float*)d_in[30];
    const float* fW8 = (const float*)d_in[31]; const float* fb8 = (const float*)d_in[32];

    const float* av   = (const float*)d_in[33];
    const float* stdv = (const float*)d_in[34];
    const float* lenp = (const float*)d_in[35];

    float*     out      = (float*)d_out;
    float*     td_table = (float*)d_ws;                           // 64 B
    _Float16*  pack     = (_Float16*)((char*)d_ws + 1024);        // 42 KB packed frags
    _Float16*  Dflat    = (_Float16*)((char*)d_ws + 65536);       // 16 MB fp16

    k_prepare<<<1, 32, 0, stream>>>(esW0, esB0, esW1, esB1,
                                    fsW0, fsB0, fsW1, fsB1,
                                    enW0, enW1, enW2,
                                    fW1, fW2, fW3, fW4,
                                    td_table, pack, out);

    k_atom_env<<<NSNAP * NPTS, 64, 0, stream>>>(pos, types, neigh,
                                                enB0, enB1, enB2,
                                                td_table, pack, lenp, Dflat);

    k_fit<<<(NSNAP * NPTS) / 16, 32, 0, stream>>>(Dflat, pack,
                                                  fb1, fb2, fb3, fb4,
                                                  fW5, fb5, fW6, fb6, fW7, fb7, fW8, fb8,
                                                  types, av, stdv, out);
}
